// Net_55319178772737
// MI455X (gfx1250) — matrix operand order repaired and checked by reading, not compiled, not run on hardware
//
#include <hip/hip_runtime.h>
#include <math.h>

// ---------------- problem constants ----------------
#define BG   128            // graphs
#define NN   512            // nodes per graph
#define DD   256            // embed dim
#define EE   8192           // edges per graph
#define NT   (BG * NN)      // 65536 total nodes
#define ET   (BG * EE)      // 1048576 total edges
#define KTOP 410            // ceil(0.8 * 512)

typedef __bf16 bf16;
typedef __attribute__((ext_vector_type(16))) bf16  v16bf;
typedef __attribute__((ext_vector_type(8)))  float v8f;

union FragU { v16bf v; unsigned u[8]; };

// ======================= CSR build =======================
__global__ void k_zero_i32(int* __restrict__ p, int n) {
    int i = blockIdx.x * blockDim.x + threadIdx.x;
    if (i < n) p[i] = 0;
}

__global__ void k_hist(const int* __restrict__ dst, int* __restrict__ cnt, int n) {
    int e = blockIdx.x * blockDim.x + threadIdx.x;
    if (e < n) atomicAdd(&cnt[dst[e]], 1);
}

// block b scans cnt[b*256 .. b*256+255] -> exclusive prefix in off, block total in bsum
__global__ __launch_bounds__(256) void k_scanA(const int* __restrict__ cnt,
                                               int* __restrict__ off,
                                               int* __restrict__ bsum) {
    __shared__ int s[256];
    int t = threadIdx.x, base = blockIdx.x * 256;
    int own = cnt[base + t];
    s[t] = own;
    for (int d = 1; d < 256; d <<= 1) {
        __syncthreads();
        int v = (t >= d) ? s[t - d] : 0;
        __syncthreads();
        s[t] += v;
    }
    __syncthreads();
    off[base + t] = s[t] - own;           // local exclusive
    if (t == 255) bsum[blockIdx.x] = s[255];
}

__global__ __launch_bounds__(256) void k_scanB(const int* __restrict__ bsum,
                                               int* __restrict__ bscan) {
    __shared__ int s[256];
    int t = threadIdx.x;
    int own = bsum[t];
    s[t] = own;
    for (int d = 1; d < 256; d <<= 1) {
        __syncthreads();
        int v = (t >= d) ? s[t - d] : 0;
        __syncthreads();
        s[t] += v;
    }
    __syncthreads();
    bscan[t] = s[t] - own;                // exclusive over block sums
}

__global__ __launch_bounds__(256) void k_scanC(int* __restrict__ off,
                                               const int* __restrict__ bscan,
                                               int* __restrict__ cur) {
    int t = threadIdx.x;
    int i = blockIdx.x * 256 + t;
    int v = off[i] + bscan[blockIdx.x];
    off[i] = v;
    cur[i] = v;
    if (blockIdx.x == 255 && t == 255) off[NT] = ET;
}

__global__ void k_scatter(const int* __restrict__ src, const int* __restrict__ dst,
                          int* __restrict__ cur, int* __restrict__ csr, int n) {
    int e = blockIdx.x * blockDim.x + threadIdx.x;
    if (e < n) {
        int p = atomicAdd(&cur[dst[e]], 1);
        csr[p] = src[e];
    }
}

// ======== weight pre-swizzle into WMMA A-fragment order ========
// GEMM computes C' = A x B with A = Wcat^T (N'=512 rows, K=256), B = h^T.
// A-fragment (16x32 bf16, ISA 7.12.2): lane l -> row n = nt*16 + (l&15); hi = l>>4;
//   u[r] packs K pair k0 = (r<4) ? 8*hi+2r : 16+8*hi+2(r-4), within 32-wide K block kt.
// Stored so each lane's 8 uints are contiguous:
//   wz[((nt*8 + kt)*32 + lane)*8 + r]
__global__ __launch_bounds__(256) void k_wswz(const float* __restrict__ Wr,
                                              const float* __restrict__ Ws,
                                              unsigned* __restrict__ wz) {
    int idx = blockIdx.x * 256 + threadIdx.x;      // 0 .. 65535  (grid = 256)
    int r    = idx & 7;
    int lane = (idx >> 3) & 31;
    int kt   = (idx >> 8) & 7;
    int nt   = idx >> 11;                          // 0..31
    int n  = nt * 16 + (lane & 15);
    int hi = lane >> 4;
    int k0 = (r < 4) ? (8 * hi + 2 * r) : (16 + 8 * hi + 2 * (r - 4));
    int k  = kt * 32 + k0;
    // A[n][k] = Wcat[k][n] = (n<256 ? Wr[k][n] : Ws[k][n-256])
    float f0 = (n < DD) ? Wr[k * DD + n]       : Ws[k * DD + n - DD];
    float f1 = (n < DD) ? Wr[(k + 1) * DD + n] : Ws[(k + 1) * DD + n - DD];
    union { bf16 h[2]; unsigned u; } pk;
    pk.h[0] = (bf16)f0;
    pk.h[1] = (bf16)f1;
    wz[idx] = pk.u;
}

// ======================= layer 1 (K=4 input features) =======================
__global__ __launch_bounds__(256) void k_layer1(const float* __restrict__ x,
                                                const int* __restrict__ off,
                                                const int* __restrict__ csr,
                                                const float* __restrict__ Wr,
                                                const float* __restrict__ Ws,
                                                const float* __restrict__ bias,
                                                float* __restrict__ h,
                                                bf16* __restrict__ hbf) {
    __shared__ float sa[4], sx[4];
    int i = blockIdx.x, t = threadIdx.x;
    if (t < 4) {
        float a = 0.f;
        int e1 = off[i + 1];
        for (int e = off[i]; e < e1; ++e) a += x[csr[e] * 4 + t];
        sa[t] = a;
    } else if (t < 8) {
        sx[t - 4] = x[i * 4 + (t - 4)];
    }
    __syncthreads();
    float o = bias[t];
#pragma unroll
    for (int k = 0; k < 4; ++k) o += sa[k] * Wr[k * DD + t] + sx[k] * Ws[k * DD + t];
    float v = fmaxf(o, 0.f);
    h[(size_t)i * DD + t]   = v;
    hbf[(size_t)i * DD + t] = (bf16)v;
}

// ======================= global max/mean pooling -> z =======================
__global__ __launch_bounds__(256) void k_pool(const float* __restrict__ h,
                                              float* __restrict__ z, int accumulate) {
    int g = blockIdx.x, d = threadIdx.x;
    const float* hp = h + (size_t)g * NN * DD + d;
    float mx = -3.402823466e38f, sm = 0.f;
    for (int n = 0; n < NN; ++n) {
        float v = hp[(size_t)n * DD];
        mx = fmaxf(mx, v);
        sm += v;
    }
    float mean = sm * (1.f / NN);
    if (accumulate) {
        z[g * 512 + d]       += mx;
        z[g * 512 + 256 + d] += mean;
    } else {
        z[g * 512 + d]       = mx;
        z[g * 512 + 256 + d] = mean;
    }
}

// =========== WMMA bf16 GEMM, LDS-free, transposed product, SW-pipelined ===========
// hW[m][n] = sum_k h[m][k] * Wcat[k][n], computed as D = A(W^T 16x32) x B(h^T 32x16).
// A fragments: 32B contiguous per lane; B: two 16B pieces. 256 thr = 8 waves; block covers 128
// node rows (wave w -> nodes blockIdx*128 + w*16 + lo); loops 32 n-tiles x 8 k-steps.
// A-fragment loads are ping-pong double-buffered so the next pair of
// global_load_b128 is in flight while the current v_wmma executes.
__global__ __launch_bounds__(256) void k_gemm_bf16(const bf16* __restrict__ hb,
                                                   const unsigned* __restrict__ wz,
                                                   float* __restrict__ hW) {
    const int wave = threadIdx.x >> 5, lane = threadIdx.x & 31;
    const int lo = lane & 15, hi = lane >> 4;
    const int m = blockIdx.x * 128 + wave * 16 + lo;      // node row (B col / D col)
    const v8f vzero = {0.f, 0.f, 0.f, 0.f, 0.f, 0.f, 0.f, 0.f};

    // preload all 8 B fragments of this node row: u[r] covers k = 32*ks + ((r<4) ? 8*hi+2r : 16+8*hi+2(r-4)) + {0,1}
    const unsigned* hrow = (const unsigned*)(hb + (size_t)m * DD);   // 128 uints
    FragU bfr[8];
#pragma unroll
    for (int ks = 0; ks < 8; ++ks) {
        const unsigned* p = hrow + ks * 16 + hi * 4;
#pragma unroll
        for (int r = 0; r < 8; ++r) bfr[ks].u[r] = p[(r < 4) ? r : (r + 4)];
    }

    // prime the A-fragment pipeline with (nt=0, ks=0)
    FragU aCur;
    {
        const unsigned* p0 = wz + lane * 8;
#pragma unroll
        for (int r = 0; r < 8; ++r) aCur.u[r] = p0[r];
    }

    for (int nt = 0; nt < 32; ++nt) {
        const unsigned* wp = wz + (size_t)nt * 2048 + lane * 8;
        if (nt < 31) __builtin_prefetch(wp + 2048, 0, 3);   // global_prefetch_b8 (near)
        v8f acc = vzero;
#pragma unroll
        for (int ks = 0; ks < 8; ++ks) {
            // issue loads for the NEXT fragment before consuming the current one
            const unsigned* pn = (ks < 7)
                ? (wp + (ks + 1) * 256)
                : (wz + (size_t)((nt < 31) ? (nt + 1) : 0) * 2048 + lane * 8);
            FragU aNxt;
#pragma unroll
            for (int r = 0; r < 8; ++r) aNxt.u[r] = pn[r];
            acc = __builtin_amdgcn_wmma_f32_16x16x32_bf16(
                false, aCur.v, false, bfr[ks].v, (short)0, acc, false, false);
            aCur = aNxt;    // pure rename under full unroll
        }
        // D: lane -> node col m; VGPR r -> feature n = nt*16 + 8*hi + r (contiguous)
        float* cp = hW + (size_t)m * 512 + nt * 16 + 8 * hi;
#pragma unroll
        for (int r = 0; r < 8; ++r) cp[r] = acc[r];
    }
}

// ============== aggregation + root + bias + relu ==============
__global__ __launch_bounds__(256) void k_agg(const float* __restrict__ hW,
                                             const int* __restrict__ off,
                                             const int* __restrict__ csr,
                                             const float* __restrict__ bias,
                                             float* __restrict__ hout,
                                             bf16* __restrict__ hbf) {
    int i = blockIdx.x, d = threadIdx.x;
    float acc = hW[(size_t)i * 512 + 256 + d] + bias[d];
    int e0 = off[i], e1 = off[i + 1];
    for (int e = e0; e < e1; ++e) acc += hW[(size_t)csr[e] * 512 + d];
    float v = fmaxf(acc, 0.f);
    hout[(size_t)i * DD + d] = v;
    hbf[(size_t)i * DD + d]  = (bf16)v;
}

// ======================= ||p|| + 1e-16 =======================
__global__ __launch_bounds__(256) void k_pnorm(const float* __restrict__ p,
                                               float* __restrict__ pden) {
    __shared__ float red[8];
    int t = threadIdx.x;
    float v = p[t];
    float s = v * v;
    for (int o = 16; o > 0; o >>= 1) s += __shfl_down(s, o);
    if ((t & 31) == 0) red[t >> 5] = s;
    __syncthreads();
    if (t == 0) {
        float tot = 0.f;
        for (int w = 0; w < 8; ++w) tot += red[w];
        pden[0] = sqrtf(tot) + 1e-16f;
    }
}

// ======================= score = (h . p) / pden, warp per node =======================
__global__ __launch_bounds__(256) void k_score(const float* __restrict__ h,
                                               const float* __restrict__ p,
                                               const float* __restrict__ pden,
                                               float* __restrict__ sc) {
    int warp = threadIdx.x >> 5, lane = threadIdx.x & 31;
    int node = blockIdx.x * 8 + warp;
    const float* hp = h + (size_t)node * DD;
    float acc = 0.f;
#pragma unroll
    for (int j = 0; j < 8; ++j) acc += hp[lane + 32 * j] * p[lane + 32 * j];
    for (int o = 16; o > 0; o >>= 1) acc += __shfl_down(acc, o);
    if (lane == 0) sc[node] = acc / pden[0];
}

// ======================= per-graph full bitonic sort (desc, stable ties) =======================
__global__ __launch_bounds__(256) void k_topk(const float* __restrict__ sc,
                                              float* __restrict__ tkv,
                                              int* __restrict__ tki) {
    __shared__ float v[NN];
    __shared__ int   id[NN];
    int g = blockIdx.x, t = threadIdx.x;
    v[t]       = sc[g * NN + t];        id[t]       = t;
    v[t + 256] = sc[g * NN + 256 + t];  id[t + 256] = t + 256;
    for (int k = 2; k <= NN; k <<= 1) {
        for (int j = k >> 1; j > 0; j >>= 1) {
            __syncthreads();
#pragma unroll
            for (int half = 0; half < 2; ++half) {
                int i = t + half * 256;
                int ixj = i ^ j;
                if (ixj > i) {
                    float vi = v[i], vj = v[ixj];
                    int   ii = id[i], ij = id[ixj];
                    bool iBefore = (vi > vj) || (vi == vj && ii < ij);
                    bool descBlk = ((i & k) == 0);
                    bool doSwap = descBlk ? !iBefore : iBefore;
                    if (doSwap) { v[i] = vj; v[ixj] = vi; id[i] = ij; id[ixj] = ii; }
                }
            }
        }
    }
    __syncthreads();
    tkv[g * NN + t]       = tanhf(v[t]);       tki[g * NN + t]       = id[t];
    tkv[g * NN + 256 + t] = tanhf(v[256 + t]); tki[g * NN + 256 + t] = id[256 + t];
}

// ======================= x3 = gmp/gap over top-K scaled rows, accumulate into z =======================
__global__ __launch_bounds__(256) void k_x3pool(const float* __restrict__ h,
                                                const float* __restrict__ tkv,
                                                const int* __restrict__ tki,
                                                float* __restrict__ z) {
    int g = blockIdx.x, d = threadIdx.x;
    float mx = -3.402823466e38f, sm = 0.f;
    for (int k = 0; k < KTOP; ++k) {
        int   i  = tki[g * NN + k];
        float tv = tkv[g * NN + k];
        float val = h[((size_t)g * NN + i) * DD + d] * tv;
        mx = fmaxf(mx, val);
        sm += val;
    }
    z[g * 512 + d]       += mx;
    z[g * 512 + 256 + d] += sm * (1.f / KTOP);
}

// ======================= final MLP: 512->256 relu ->128 relu ->1 sigmoid =======================
__global__ __launch_bounds__(256) void k_mlp(const float* __restrict__ z,
                                             const float* __restrict__ W1, const float* __restrict__ c1,
                                             const float* __restrict__ W2, const float* __restrict__ c2,
                                             const float* __restrict__ W3, const float* __restrict__ c3,
                                             float* __restrict__ out) {
    __shared__ float zr[512], a1[256], a2[128], red[8];
    int g = blockIdx.x, t = threadIdx.x;
    zr[t]       = z[g * 512 + t];
    zr[t + 256] = z[g * 512 + 256 + t];
    __syncthreads();
    float acc = c1[t];
    for (int k = 0; k < 512; ++k) acc += zr[k] * W1[k * 256 + t];
    a1[t] = fmaxf(acc, 0.f);
    __syncthreads();
    if (t < 128) {
        float a = c2[t];
        for (int k = 0; k < 256; ++k) a += a1[k] * W2[k * 128 + t];
        a2[t] = fmaxf(a, 0.f);
    }
    __syncthreads();
    float p = (t < 128) ? a2[t] * W3[t] : 0.f;
    for (int o = 16; o > 0; o >>= 1) p += __shfl_down(p, o);
    if ((t & 31) == 0) red[t >> 5] = p;
    __syncthreads();
    if (t == 0) {
        float s = c3[0];
        for (int w = 0; w < 8; ++w) s += red[w];
        out[g] = 1.f / (1.f + __expf(-s));
    }
}

// ======================= driver =======================
extern "C" void kernel_launch(void* const* d_in, const int* in_sizes, int n_in,
                              void* d_out, int out_size, void* d_ws, size_t ws_size,
                              hipStream_t stream) {
    (void)in_sizes; (void)n_in; (void)out_size; (void)ws_size;
    const float* x   = (const float*)d_in[0];
    const int*   src = (const int*)d_in[1];
    const int*   dst = (const int*)d_in[2];
    const float* Wr1 = (const float*)d_in[3];
    const float* Ws1 = (const float*)d_in[4];
    const float* b1  = (const float*)d_in[5];
    const float* Wr2 = (const float*)d_in[6];
    const float* Ws2 = (const float*)d_in[7];
    const float* b2  = (const float*)d_in[8];
    const float* Wr3 = (const float*)d_in[9];
    const float* Ws3 = (const float*)d_in[10];
    const float* b3  = (const float*)d_in[11];
    const float* pa  = (const float*)d_in[12];
    const float* W1  = (const float*)d_in[13];
    const float* c1  = (const float*)d_in[14];
    const float* W2  = (const float*)d_in[15];
    const float* c2  = (const float*)d_in[16];
    const float* W3  = (const float*)d_in[17];
    const float* c3  = (const float*)d_in[18];
    float* out = (float*)d_out;

    // workspace carve-out (all 256B aligned)
    char* w = (char*)d_ws;
    auto carve = [&](size_t bytes) -> char* {
        char* p = w;
        w += (bytes + 255) & ~(size_t)255;
        return p;
    };
    float*    hN  = (float*)   carve((size_t)NT * DD * 4);   // node features f32 (64 MB)
    bf16*     hBF = (bf16*)    carve((size_t)NT * DD * 2);   // node features bf16 (32 MB)
    float*    hW  = (float*)   carve((size_t)NT * 512 * 4);  // [h@Wr | h@Ws] (128 MB)
    unsigned* wz2 = (unsigned*)carve((size_t)65536 * 4);     // swizzled weights (256 KB)
    unsigned* wz3 = (unsigned*)carve((size_t)65536 * 4);
    int*   cnt  = (int*)  carve((size_t)NT * 4);
    int*   off  = (int*)  carve((size_t)(NT + 1) * 4);
    int*   cur  = (int*)  carve((size_t)NT * 4);
    int*   csr  = (int*)  carve((size_t)ET * 4);
    int*   bsum = (int*)  carve(256 * 4);
    int*   bscn = (int*)  carve(256 * 4);
    float* z    = (float*)carve((size_t)BG * 512 * 4);
    float* scb  = (float*)carve((size_t)NT * 4);
    float* tkv  = (float*)carve((size_t)BG * NN * 4);
    int*   tki  = (int*)  carve((size_t)BG * NN * 4);
    float* pden = (float*)carve(256);

    // ---- CSR by destination (once per call) ----
    k_zero_i32<<<NT / 256, 256, 0, stream>>>(cnt, NT);
    k_hist<<<ET / 256, 256, 0, stream>>>(dst, cnt, ET);
    k_scanA<<<256, 256, 0, stream>>>(cnt, off, bsum);
    k_scanB<<<1, 256, 0, stream>>>(bsum, bscn);
    k_scanC<<<256, 256, 0, stream>>>(off, bscn, cur);
    k_scatter<<<ET / 256, 256, 0, stream>>>(src, dst, cur, csr, ET);

    // ---- weight prep (fragment-swizzled bf16) ----
    k_wswz<<<256, 256, 0, stream>>>(Wr2, Ws2, wz2);
    k_wswz<<<256, 256, 0, stream>>>(Wr3, Ws3, wz3);
    k_pnorm<<<1, 256, 0, stream>>>(pa, pden);

    // ---- layer 1 (fp32 scalar, K=4) + pooling x1 ----
    k_layer1<<<NT, 256, 0, stream>>>(x, off, csr, Wr1, Ws1, b1, hN, hBF);
    k_pool<<<BG, 256, 0, stream>>>(hN, z, 0);

    // ---- layer 2: WMMA GEMM -> gather-agg -> pooling x2 ----
    k_gemm_bf16<<<NT / 128, 256, 0, stream>>>(hBF, wz2, hW);
    k_agg<<<NT, 256, 0, stream>>>(hW, off, csr, b2, hN, hBF);
    k_pool<<<BG, 256, 0, stream>>>(hN, z, 1);

    // ---- layer 3: WMMA GEMM -> gather-agg ----
    k_gemm_bf16<<<NT / 128, 256, 0, stream>>>(hBF, wz3, hW);
    k_agg<<<NT, 256, 0, stream>>>(hW, off, csr, b3, hN, hBF);

    // ---- TopK pooling -> x3 ----
    k_score<<<NT / 8, 256, 0, stream>>>(hN, pa, pden, scb);
    k_topk<<<BG, 256, 0, stream>>>(scb, tkv, tki);
    k_x3pool<<<BG, 256, 0, stream>>>(hN, tkv, tki, z);

    // ---- final MLP ----
    k_mlp<<<BG, 256, 0, stream>>>(z, W1, c1, W2, c2, W3, c3, out);
}
